// Fusion_12867722019048
// MI455X (gfx1250) — matrix operand order repaired and checked by reading, not compiled, not run on hardware
//
#include <hip/hip_runtime.h>
#include <hip/hip_bf16.h>
#include <math.h>

typedef __attribute__((ext_vector_type(16))) _Float16 v16h;
typedef __attribute__((ext_vector_type(8)))  float    v8f;

#define HPAD 17      // 16x16 f32 staging padded to kill LDS bank conflicts
#define YSTR 136     // 128 f16 + 8 pad -> row stride 272B = 68 dwords

// f16 weight cache layout inside d_ws (element offsets, _Float16 units)
#define OFF_WA   0u
#define OFF_WV   1024u
#define OFF_WL   2048u
#define OFF_WAP  3072u
#define OFF_WVP  4096u
#define OFF_WLP  5120u
#define OFF_WF1  6144u            // 64x256
#define OFF_WFP1 (6144u + 16384u) // 64x256
#define OFF_WNG  (6144u + 32768u) // 64x128
#define WH_TOTAL (6144u + 32768u + 8192u)  // 47104 halves = 94208 bytes

struct WaveScratch {
  float    h[3][16][HPAD];   // a_h / v_h / l_h (reused across branches)
  _Float16 y[16][YSTR];      // gated concat, f16, WMMA-A staging
};

// fast sigmoid: v_exp_f32 + v_rcp_f32 (avoid the IEEE-div expansion)
__device__ __forceinline__ float sigm(float x) {
  return __builtin_amdgcn_rcpf(1.0f + __expf(-x));
}

// 16-bit A-matrix 16x32 layout, contiguous K order (matches B):
// lanes 0-15 (half=0): idx 0..15 -> K=0..15
// lanes 16-31(half=1): idx 0..15 -> K=16..31
__device__ __forceinline__ int a_k(int idx, int h) {
  return h * 16 + idx;
}

__device__ __forceinline__ v16h load_a_frag_g(const float* xr, int h, int kbase) {
  v16h a;
#pragma unroll
  for (int idx = 0; idx < 16; ++idx) a[idx] = (_Float16)xr[kbase + a_k(idx, h)];
  return a;
}

// B-matrix 32x16 f16 layout: lane (col n = lane&15, half h) holds K = h*16 + idx.
// B[k][n] = W[n][k]  (Y = X @ W^T).  Weights are pre-converted f16 -> pure b128 loads.
__device__ __forceinline__ v16h load_b_frag_h(const _Float16* W, int ld, int orow, int kbase, int h) {
  const _Float16* wr = W + (size_t)orow * ld + kbase + h * 16;
  v16h b;
#pragma unroll
  for (int idx = 0; idx < 16; ++idx) b[idx] = wr[idx];
  return b;
}

__device__ __forceinline__ v8f wmma16(v16h a, v16h b, v8f c) {
  return __builtin_amdgcn_wmma_f32_16x16x32_f16(false, a, false, b, (short)0, c, false, false);
}

// [16,64] @ W[16,64]^T + bias -> sigmoid -> LDS [16][16] f32 (C layout)
__device__ __forceinline__ void project_store(const float* X, int rowBase,
                                              const _Float16* W, const float* bias,
                                              float (*dst)[HPAD], int lo, int h) {
  const float* xr = X + (size_t)(rowBase + lo) * 64;
  v16h a0 = load_a_frag_g(xr, h, 0);
  v16h a1 = load_a_frag_g(xr, h, 32);
  v16h b0 = load_b_frag_h(W, 64, lo, 0, h);
  v16h b1 = load_b_frag_h(W, 64, lo, 32, h);
  v8f c = {};
  c = wmma16(a0, b0, c);
  c = wmma16(a1, b1, c);
  float bb = bias[lo];
#pragma unroll
  for (int i = 0; i < 8; ++i) dst[h * 8 + i][lo] = sigm(c[i] + bb);
}

// Trimodal fusion (collapsed pooling of positive products) + [16,256]x[256,64] GEMM + leaky_relu.
__device__ __forceinline__ void fusion_branch(const WaveScratch* ws, int lo, int h,
                                              const _Float16* Wf, const float* bf, v8f fout[4]) {
  float A2[8], V2[8], L2[8];
#pragma unroll
  for (int i = 0; i < 8; ++i) {
    A2[i] = fmaxf(ws->h[0][lo][2 * i], ws->h[0][lo][2 * i + 1]);
    V2[i] = fmaxf(ws->h[1][lo][2 * i], ws->h[1][lo][2 * i + 1]);
    L2[i] = fmaxf(ws->h[2][lo][2 * i], ws->h[2][lo][2 * i + 1]);
  }
  float VV[4];
#pragma unroll
  for (int u = 0; u < 4; ++u) VV[u] = fmaxf(V2[2 * u], V2[2 * u + 1]);

  // f16 factor tables so the 128 products pack as v_pk_mul_f16
  _Float16 Ph[32], Lh[8];
#pragma unroll
  for (int p = 0; p < 32; ++p) Ph[p] = (_Float16)(A2[p >> 2] * VV[p & 3]);
#pragma unroll
  for (int q = 0; q < 8; ++q) Lh[q] = (_Float16)L2[q];

  // Synthesize the 16x256 fusion matrix directly in A-fragment layout: t[k] = P[k>>3]*L2[k&7]
  v16h afr[8];
#pragma unroll
  for (int kf = 0; kf < 8; ++kf) {
#pragma unroll
    for (int idx = 0; idx < 16; ++idx) {
      int k = kf * 32 + a_k(idx, h);
      afr[kf][idx] = Ph[k >> 3] * Lh[k & 7];
    }
  }
#pragma unroll
  for (int nt = 0; nt < 4; ++nt) {
    v8f acc = {};
#pragma unroll
    for (int kf = 0; kf < 8; ++kf)
      acc = wmma16(afr[kf], load_b_frag_h(Wf, 256, nt * 16 + lo, kf * 32, h), acc);
    float bb = bf[nt * 16 + lo];
#pragma unroll
    for (int i = 0; i < 8; ++i) {
      float x = acc[i] + bb;
      acc[i] = x > 0.0f ? x : 0.01f * x;   // leaky_relu, slope 0.01
    }
    fout[nt] = acc;
  }
}

// One-time f32 -> f16 weight conversion into d_ws
__global__ void cvt_weights_kernel(const float* src, _Float16* dst, int n) {
  int i = blockIdx.x * 256 + threadIdx.x;
  if (i < n) dst[i] = (_Float16)src[i];
}

__global__ __launch_bounds__(256) void fused_trimodal_kernel(
    const float* a, const float* v, const float* l,
    const float* pa, const float* pv, const float* pl, const float* mean,
    const _Float16* wh,
    const float* ba, const float* bv, const float* bl,
    const float* bap, const float* bvp, const float* blp,
    const float* bf1, const float* bfp1,
    const float* bng, const float* rm1, const float* rm2,
    float* out) {
  __shared__ WaveScratch smem[8];
  const int wave = threadIdx.x >> 5;
  const int lane = threadIdx.x & 31;
  const int lo = lane & 15;        // row for A, column for B/C/D
  const int h  = lane >> 4;        // half-wave
  const int rowBase = blockIdx.x * 128 + wave * 16;
  WaveScratch* ws = &smem[wave];

  // ---- branch 1: projections -> fusion1 ----
  project_store(a, rowBase, wh + OFF_WA, ba, ws->h[0], lo, h);
  project_store(v, rowBase, wh + OFF_WV, bv, ws->h[1], lo, h);
  project_store(l, rowBase, wh + OFF_WL, bl, ws->h[2], lo, h);
  __syncthreads();
  v8f f1[4];
  fusion_branch(ws, lo, h, wh + OFF_WF1, bf1, f1);
  __syncthreads();

  // ---- branch 2: projections -> fusion2 ----
  project_store(pa, rowBase, wh + OFF_WAP, bap, ws->h[0], lo, h);
  project_store(pv, rowBase, wh + OFF_WVP, bvp, ws->h[1], lo, h);
  project_store(pl, rowBase, wh + OFF_WLP, blp, ws->h[2], lo, h);
  __syncthreads();
  v8f f2[4];
  fusion_branch(ws, lo, h, wh + OFF_WFP1, bfp1, f2);

  // ---- normGate 1: y = gate([fusion2, mean], rm1), staged as f16 A operand ----
#pragma unroll
  for (int nt = 0; nt < 4; ++nt) {
    int col = nt * 16 + lo;
    float rm = rm1[col];
#pragma unroll
    for (int i = 0; i < 8; ++i) {
      float x = f2[nt][i];
      ws->y[h * 8 + i][col] = (_Float16)(x * sigm(fabsf(x - rm)));
    }
  }
  {
    const float* mr = mean + (size_t)(rowBase + lo) * 64 + h * 32;
#pragma unroll
    for (int j = 0; j < 32; ++j) {
      int c = h * 32 + j;
      float x = mr[j];
      ws->y[lo][64 + c] = (_Float16)(x * sigm(fabsf(x - rm1[64 + c])));
    }
  }
  __syncthreads();

  // ---- out1 = y @ Wng^T + bng  (K=128, N=64) ----
  v16h ya[4];
#pragma unroll
  for (int kf = 0; kf < 4; ++kf) {
#pragma unroll
    for (int idx = 0; idx < 16; ++idx)
      ya[kf][idx] = ws->y[lo][kf * 32 + a_k(idx, h)];
  }
#pragma unroll
  for (int nt = 0; nt < 4; ++nt) {
    v8f acc = {};
#pragma unroll
    for (int kf = 0; kf < 4; ++kf)
      acc = wmma16(ya[kf], load_b_frag_h(wh + OFF_WNG, 128, nt * 16 + lo, kf * 32, h), acc);
    int col = nt * 16 + lo;
    float bb  = bng[col];
    float r2a = rm2[col];
    float r2b = rm2[64 + col];
#pragma unroll
    for (int i = 0; i < 8; ++i) {
      size_t row = (size_t)rowBase + h * 8 + i;
      float x = acc[i] + bb;                       // out1 column `col`
      out[row * 128 + col] = x * sigm(fabsf(x - r2a));
      float z = f1[nt][i];                         // fusion1 column `col`
      out[row * 128 + 64 + col] = z * sigm(fabsf(z - r2b));
    }
  }
}

extern "C" void kernel_launch(void* const* d_in, const int* in_sizes, int n_in,
                              void* d_out, int out_size, void* d_ws, size_t ws_size,
                              hipStream_t stream) {
  (void)in_sizes; (void)n_in; (void)ws_size; (void)out_size;
  const float* a    = (const float*)d_in[0];
  const float* v    = (const float*)d_in[1];
  const float* l    = (const float*)d_in[2];
  const float* pa   = (const float*)d_in[3];
  const float* pv   = (const float*)d_in[4];
  const float* pl   = (const float*)d_in[5];
  const float* mean = (const float*)d_in[6];
  const float* Wa   = (const float*)d_in[7];
  const float* ba   = (const float*)d_in[8];
  const float* Wv   = (const float*)d_in[9];
  const float* bv   = (const float*)d_in[10];
  const float* Wl   = (const float*)d_in[11];
  const float* bl   = (const float*)d_in[12];
  const float* Wap  = (const float*)d_in[13];
  const float* bap  = (const float*)d_in[14];
  const float* Wvp  = (const float*)d_in[15];
  const float* bvp  = (const float*)d_in[16];
  const float* Wlp  = (const float*)d_in[17];
  const float* blp  = (const float*)d_in[18];
  const float* Wf1  = (const float*)d_in[19];
  const float* bf1  = (const float*)d_in[20];
  const float* Wfp1 = (const float*)d_in[21];
  const float* bfp1 = (const float*)d_in[22];
  const float* Wng  = (const float*)d_in[23];
  const float* bng  = (const float*)d_in[24];
  const float* rm1  = (const float*)d_in[25];
  const float* rm2  = (const float*)d_in[26];
  float* out = (float*)d_out;

  _Float16* wh = (_Float16*)d_ws;

  // one-time (per launch) weight f32->f16 conversion; tiny vs. the 144 MB main traffic
  cvt_weights_kernel<<<(1024 + 255) / 256, 256, 0, stream>>>(Wa,   wh + OFF_WA,   1024);
  cvt_weights_kernel<<<(1024 + 255) / 256, 256, 0, stream>>>(Wv,   wh + OFF_WV,   1024);
  cvt_weights_kernel<<<(1024 + 255) / 256, 256, 0, stream>>>(Wl,   wh + OFF_WL,   1024);
  cvt_weights_kernel<<<(1024 + 255) / 256, 256, 0, stream>>>(Wap,  wh + OFF_WAP,  1024);
  cvt_weights_kernel<<<(1024 + 255) / 256, 256, 0, stream>>>(Wvp,  wh + OFF_WVP,  1024);
  cvt_weights_kernel<<<(1024 + 255) / 256, 256, 0, stream>>>(Wlp,  wh + OFF_WLP,  1024);
  cvt_weights_kernel<<<(16384 + 255) / 256, 256, 0, stream>>>(Wf1,  wh + OFF_WF1,  16384);
  cvt_weights_kernel<<<(16384 + 255) / 256, 256, 0, stream>>>(Wfp1, wh + OFF_WFP1, 16384);
  cvt_weights_kernel<<<(8192 + 255) / 256, 256, 0, stream>>>(Wng,  wh + OFF_WNG,  8192);

  dim3 grid(65536 / 128);   // 8 waves/block, 16 rows per wave
  dim3 block(256);
  fused_trimodal_kernel<<<grid, block, 0, stream>>>(
      a, v, l, pa, pv, pl, mean, wh,
      ba, bv, bl, bap, bvp, blp, bf1, bfp1, bng, rm1, rm2, out);
}
